// MultiHeadSelfAttention_57501022159272
// MI455X (gfx1250) — compile-verified, matrix operand order checked by reading, not run on hardware
//
#include <hip/hip_runtime.h>
#include <hip/hip_bf16.h>
#include <math.h>

typedef __attribute__((ext_vector_type(16))) _Float16 v16h;
typedef __attribute__((ext_vector_type(8)))  _Float16 v8h;
typedef __attribute__((ext_vector_type(4)))  _Float16 v4h;
typedef __attribute__((ext_vector_type(8)))  float    v8f;
typedef __attribute__((ext_vector_type(4)))  unsigned int v4u;
typedef __attribute__((ext_vector_type(8)))  int      v8i;
typedef __attribute__((ext_vector_type(4)))  int      v4i;

#define D_MODEL 2048
#define N_HEADS 16
#define D_HEAD  128
#define T_SEQ   2048
#define B_SZ    2
#define QKV_COLS (3 * D_MODEL)

// 16-bit A/B fragment (ISA 7.12.2, wave32): per lane the 16 halves are two
// contiguous K-runs: [8*hi, 8*hi+7] and [16+8*hi, 16+8*hi+7].
__device__ __forceinline__ v16h ldfrag(const _Float16* p) {
    v8h a = *(const v8h*)p;          // K = base .. base+7
    v8h b = *(const v8h*)(p + 16);   // K = base+16 .. base+23
    return __builtin_shufflevector(a, b, 0, 1, 2, 3, 4, 5, 6, 7,
                                   8, 9, 10, 11, 12, 13, 14, 15);
}

// ---------------------------------------------------------------------------
// TDM: async 2D tile load (global -> LDS), D# per ISA 8.3/8.4.
// data_size=2 bytes; tile rows stored contiguously in LDS (row-major,
// tile_dim0 halves per row). Tracked by TENSORcnt.
// Toolchain probe: 6-arg builtin (g0, g1, g2, g3, g4, cpol).
// ---------------------------------------------------------------------------
__device__ __forceinline__ void tdm_load_2d(void* lds_dst, const _Float16* gsrc,
                                            unsigned dim0, unsigned dim1,
                                            unsigned tile0, unsigned tile1,
                                            unsigned long long stride0) {
    const unsigned long long ga = (unsigned long long)(uintptr_t)gsrc;
    v4u g0;
    g0[0] = 1u;                                   // count=1, user descriptor
    g0[1] = (unsigned)(uintptr_t)lds_dst;         // LDS byte address (low 32 bits)
    g0[2] = (unsigned)ga;                         // global_addr[31:0]
    g0[3] = (unsigned)((ga >> 32) & 0x01FFFFFFu)  // global_addr[56:32]
            | 0x80000000u;                        // type=2 ("image")
    v8i g1;
    g1[0] = (int)(1u << 16);                      // data_size=1 -> 2 bytes
    g1[1] = (int)((dim0 & 0xFFFFu) << 16);        // tensor_dim0[15:0] @ bits 63:48
    g1[2] = (int)((dim0 >> 16) | ((dim1 & 0xFFFFu) << 16));   // dim0 hi | dim1 lo
    g1[3] = (int)((dim1 >> 16) | (tile0 << 16));  // dim1 hi | tile_dim0
    g1[4] = (int)(tile1 & 0xFFFFu);               // tile_dim1 (tile_dim2 = 0)
    g1[5] = (int)(unsigned)stride0;               // tensor_dim0_stride[31:0]
    g1[6] = (int)(unsigned)(stride0 >> 32);       // stride0[47:32] (stride1 lo16 = 0)
    g1[7] = 0;
    v4i z4 = {0, 0, 0, 0};
    v8i z8 = {0, 0, 0, 0, 0, 0, 0, 0};
    __builtin_amdgcn_tensor_load_to_lds(g0, g1, z4, z4, z8, 0);
}

// ---------------------------------------------------------------------------
// Prep kernels: f32->f16 copy of x; f32 [K,N] -> f16 [N,K] weight transposes.
// ---------------------------------------------------------------------------
__global__ void cvt_f16_kernel(const float* __restrict__ src,
                               _Float16* __restrict__ dst, int n4) {
    int i = blockIdx.x * blockDim.x + threadIdx.x;
    if (i >= n4) return;
    const float4 f = ((const float4*)src)[i];
    v4h h;
    h[0] = (_Float16)f.x; h[1] = (_Float16)f.y;
    h[2] = (_Float16)f.z; h[3] = (_Float16)f.w;
    ((v4h*)dst)[i] = h;
}

__global__ void transpose_f16_kernel(const float* __restrict__ W,
                                     _Float16* __restrict__ WT,
                                     int K, int N) {
    size_t idx = (size_t)blockIdx.x * blockDim.x + threadIdx.x;
    if (idx >= (size_t)K * N) return;
    const int k = (int)(idx / N);
    const int n = (int)(idx % N);
    WT[(size_t)n * K + k] = (_Float16)W[idx];
}

// ---------------------------------------------------------------------------
// Kernel 1: qkv = x @ Wqkv + bqkv, RoPE on q/k. Block = 4 waves = 64x64 tile.
// B tile (64 cols x 32 K) staged by TDM into double-buffered LDS, shared by
// all 4 waves; A fragments software-pipelined from global.
// ---------------------------------------------------------------------------
__global__ void qkv_rope_kernel(const _Float16* __restrict__ xh,
                                const _Float16* __restrict__ wqT,  // [6144,2048]
                                const float* __restrict__ bqkv,
                                _Float16* __restrict__ qh,
                                _Float16* __restrict__ kh,
                                _Float16* __restrict__ vT) {
    __shared__ _Float16 bstage[2][64 * 32];

    const int wave = threadIdx.x >> 5;
    const int l = threadIdx.x & 31;
    const int lane16 = l & 15;
    const int hi = l >> 4;
    const int row0 = (blockIdx.y * 4 + wave) * 16;
    const int col0 = blockIdx.x * 64;
    const int nIter = D_MODEL / 32;

    const _Float16* arow = xh + (size_t)(row0 + lane16) * D_MODEL + 8 * hi;
    const _Float16* wbase = wqT + (size_t)col0 * D_MODEL;

    if (wave == 0)   // one wave issues the TDM descriptor (EXEC ignored by TDM)
        tdm_load_2d(bstage[0], wbase, D_MODEL, QKV_COLS - col0, 32, 64, D_MODEL);

    v16h af = ldfrag(arow);
    v8f acc[4] = {};

    for (int i = 0; i < nIter; ++i) {
        __syncthreads();                       // buf[(i+1)&1] free for overwrite
        if (wave == 0) {
            if (i + 1 < nIter) {
                tdm_load_2d(bstage[(i + 1) & 1], wbase + (i + 1) * 32,
                            D_MODEL, QKV_COLS - col0, 32, 64, D_MODEL);
                __builtin_amdgcn_s_wait_tensorcnt(1);   // tile i arrived
            } else {
                __builtin_amdgcn_s_wait_tensorcnt(0);
            }
        }
        __syncthreads();                       // tile i visible to all waves

        const v16h af_next = (i + 1 < nIter) ? ldfrag(arow + (i + 1) * 32) : af;
        const _Float16* bl = bstage[i & 1];
#pragma unroll
        for (int c = 0; c < 4; ++c) {
            const v16h bf = ldfrag(bl + (c * 16 + lane16) * 32 + 8 * hi);
            acc[c] = __builtin_amdgcn_wmma_f32_16x16x32_f16(false, af, false, bf,
                                                            (short)0, acc[c],
                                                            false, false);
        }
        af = af_next;
    }

#pragma unroll
    for (int c = 0; c < 4; ++c) {
        const int col_b = col0 + c * 16 + lane16;
        const float bias = bqkv[col_b];
        const int which = col_b >> 11;          // 0=q, 1=k, 2=v (uniform per tile)
        const int rem = col_b & (D_MODEL - 1);
        const int head = rem >> 7;
        const int d = rem & (D_HEAD - 1);
#pragma unroll
        for (int r = 0; r < 8; ++r) {
            const int row = row0 + r + 8 * hi;
            const int bidx = row >> 11;
            const int t = row & (T_SEQ - 1);
            float val = acc[c][r] + bias;
            if (which < 2) {
                const float partner = __shfl_xor(val, 1, 32);  // rotate pair col^1
                const float rot = (d & 1) ? partner : -partner;
                const float invf = __expf((float)(d & 63) * (-9.210340371976184f / 64.0f));
                const float ang = (float)t * invf;
                float sv, cv;
                __sincosf(ang, &sv, &cv);
                val = val * cv + rot * sv;
                _Float16* dst = (which == 0) ? qh : kh;
                dst[(((size_t)bidx * N_HEADS + head) * T_SEQ + t) * D_HEAD + d] =
                    (_Float16)val;
            } else {
                // V transposed: [B,H,D,T] so P@V B-fragments are K-contiguous
                vT[(((size_t)bidx * N_HEADS + head) * D_HEAD + d) * T_SEQ + t] =
                    (_Float16)val;
            }
        }
    }
}

// ---------------------------------------------------------------------------
// Kernel 2: causal flash attention per (b, h, 16-row q tile), all WMMA.
// ---------------------------------------------------------------------------
__global__ void attention_kernel(const _Float16* __restrict__ qh,
                                 const _Float16* __restrict__ kh,
                                 const _Float16* __restrict__ vT,
                                 _Float16* __restrict__ yh) {
    __shared__ _Float16 pbuf[16 * 32];   // C->A relayout staging for P

    const int l = threadIdx.x;
    const int lane16 = l & 15;
    const int hi = l >> 4;
    const int bh = blockIdx.y;           // b*16 + h
    const int b = bh >> 4;
    const int h = bh & 15;
    const int q0 = blockIdx.x * 16;

    const _Float16* Qp = qh + (size_t)bh * T_SEQ * D_HEAD;
    const _Float16* Kp = kh + (size_t)bh * T_SEQ * D_HEAD;
    const _Float16* Vp = vT + (size_t)bh * D_HEAD * T_SEQ;

    v16h qf[4];
#pragma unroll
    for (int c = 0; c < 4; ++c)
        qf[c] = ldfrag(Qp + (size_t)(q0 + lane16) * D_HEAD + c * 32 + 8 * hi);

    v8f o[8];
#pragma unroll
    for (int c = 0; c < 8; ++c)
#pragma unroll
        for (int r = 0; r < 8; ++r) o[c][r] = 0.0f;

    float mrow[8], lrow[8];
#pragma unroll
    for (int r = 0; r < 8; ++r) { mrow[r] = -1e30f; lrow[r] = 0.0f; }

    const float sm_scale = 0.08838834764831845f;  // 1/sqrt(128)

    for (int kb = 0; kb < q0 + 16; kb += 32) {    // 32 keys per chunk
        v8f st[2];
#pragma unroll
        for (int s = 0; s < 2; ++s) {
            const int n0 = kb + s * 16;
            v8f sa = {};
#pragma unroll
            for (int c = 0; c < 4; ++c) {
                const v16h bfk = ldfrag(Kp + (size_t)(n0 + lane16) * D_HEAD
                                            + c * 32 + 8 * hi);
                sa = __builtin_amdgcn_wmma_f32_16x16x32_f16(false, qf[c], false, bfk,
                                                            (short)0, sa, false, false);
            }
#pragma unroll
            for (int r = 0; r < 8; ++r) {
                const int row = q0 + r + 8 * hi;
                const int key = n0 + lane16;
                st[s][r] = (key <= row) ? sa[r] * sm_scale : -1e30f;
            }
        }

        float scale_old[8];
#pragma unroll
        for (int r = 0; r < 8; ++r) {
            float lm = fmaxf(st[0][r], st[1][r]);
            lm = fmaxf(lm, __shfl_xor(lm, 8, 32));
            lm = fmaxf(lm, __shfl_xor(lm, 4, 32));
            lm = fmaxf(lm, __shfl_xor(lm, 2, 32));
            lm = fmaxf(lm, __shfl_xor(lm, 1, 32));
            const float mnew = fmaxf(mrow[r], lm);
            scale_old[r] = __expf(mrow[r] - mnew);
            const float p0 = __expf(st[0][r] - mnew);
            const float p1 = __expf(st[1][r] - mnew);
            st[0][r] = p0; st[1][r] = p1;
            float rs = p0 + p1;
            rs += __shfl_xor(rs, 8, 32);
            rs += __shfl_xor(rs, 4, 32);
            rs += __shfl_xor(rs, 2, 32);
            rs += __shfl_xor(rs, 1, 32);
            lrow[r] = lrow[r] * scale_old[r] + rs;
            mrow[r] = mnew;
        }
#pragma unroll
        for (int c = 0; c < 8; ++c)
#pragma unroll
            for (int r = 0; r < 8; ++r) o[c][r] *= scale_old[r];

        __syncthreads();
#pragma unroll
        for (int s = 0; s < 2; ++s)
#pragma unroll
            for (int r = 0; r < 8; ++r)
                pbuf[(r + 8 * hi) * 32 + s * 16 + lane16] = (_Float16)st[s][r];
        __syncthreads();
        const v16h pf = ldfrag(&pbuf[lane16 * 32 + 8 * hi]);  // 2 x ds_load_b128

#pragma unroll
        for (int c = 0; c < 8; ++c) {
            const v16h bfv = ldfrag(Vp + (size_t)(c * 16 + lane16) * T_SEQ
                                        + kb + 8 * hi);
            o[c] = __builtin_amdgcn_wmma_f32_16x16x32_f16(false, pf, false, bfv,
                                                          (short)0, o[c], false, false);
        }
    }

#pragma unroll
    for (int r = 0; r < 8; ++r) {
        const int t = q0 + r + 8 * hi;
        const float invl = 1.0f / lrow[r];
#pragma unroll
        for (int c = 0; c < 8; ++c) {
            yh[((size_t)(b * T_SEQ + t)) * D_MODEL + h * D_HEAD + c * 16 + lane16] =
                (_Float16)(o[c][r] * invl);
        }
    }
}

// ---------------------------------------------------------------------------
// Kernel 3: out = y @ Wo + bo. Same TDM-staged structure as kernel 1.
// ---------------------------------------------------------------------------
__global__ void outproj_kernel(const _Float16* __restrict__ yh,
                               const _Float16* __restrict__ woT,
                               const float* __restrict__ bo,
                               float* __restrict__ out) {
    __shared__ _Float16 bstage[2][64 * 32];

    const int wave = threadIdx.x >> 5;
    const int l = threadIdx.x & 31;
    const int lane16 = l & 15;
    const int hi = l >> 4;
    const int row0 = (blockIdx.y * 4 + wave) * 16;
    const int col0 = blockIdx.x * 64;
    const int nIter = D_MODEL / 32;

    const _Float16* arow = yh + (size_t)(row0 + lane16) * D_MODEL + 8 * hi;
    const _Float16* wbase = woT + (size_t)col0 * D_MODEL;

    if (wave == 0)
        tdm_load_2d(bstage[0], wbase, D_MODEL, D_MODEL - col0, 32, 64, D_MODEL);

    v16h af = ldfrag(arow);
    v8f acc[4] = {};

    for (int i = 0; i < nIter; ++i) {
        __syncthreads();
        if (wave == 0) {
            if (i + 1 < nIter) {
                tdm_load_2d(bstage[(i + 1) & 1], wbase + (i + 1) * 32,
                            D_MODEL, D_MODEL - col0, 32, 64, D_MODEL);
                __builtin_amdgcn_s_wait_tensorcnt(1);
            } else {
                __builtin_amdgcn_s_wait_tensorcnt(0);
            }
        }
        __syncthreads();

        const v16h af_next = (i + 1 < nIter) ? ldfrag(arow + (i + 1) * 32) : af;
        const _Float16* bl = bstage[i & 1];
#pragma unroll
        for (int c = 0; c < 4; ++c) {
            const v16h bf = ldfrag(bl + (c * 16 + lane16) * 32 + 8 * hi);
            acc[c] = __builtin_amdgcn_wmma_f32_16x16x32_f16(false, af, false, bf,
                                                            (short)0, acc[c],
                                                            false, false);
        }
        af = af_next;
    }

#pragma unroll
    for (int c = 0; c < 4; ++c) {
        const float bias = bo[col0 + c * 16 + lane16];
#pragma unroll
        for (int r = 0; r < 8; ++r) {
            const int row = row0 + r + 8 * hi;
            out[(size_t)row * D_MODEL + col0 + c * 16 + lane16] = acc[c][r] + bias;
        }
    }
}

// ---------------------------------------------------------------------------
extern "C" void kernel_launch(void* const* d_in, const int* in_sizes, int n_in,
                              void* d_out, int out_size, void* d_ws, size_t ws_size,
                              hipStream_t stream) {
    const float* x    = (const float*)d_in[0];
    // d_in[1] = mask (unused; causal mask applied analytically)
    const float* Wqkv = (const float*)d_in[2];
    const float* bqkv = (const float*)d_in[3];
    const float* Wo   = (const float*)d_in[4];
    const float* bo   = (const float*)d_in[5];
    float* out = (float*)d_out;

    const size_t n_x   = (size_t)B_SZ * T_SEQ * D_MODEL;
    const size_t n_wq  = (size_t)D_MODEL * QKV_COLS;
    const size_t n_wo  = (size_t)D_MODEL * D_MODEL;
    const size_t n_per = (size_t)B_SZ * N_HEADS * T_SEQ * D_HEAD;

    _Float16* xh  = (_Float16*)d_ws;
    _Float16* wqT = xh + n_x;
    _Float16* woT = wqT + n_wq;
    _Float16* qh  = woT + n_wo;
    _Float16* kh  = qh + n_per;
    _Float16* vT  = kh + n_per;
    _Float16* yh  = vT + n_per;

    cvt_f16_kernel<<<(int)((n_x / 4 + 255) / 256), 256, 0, stream>>>(x, xh, (int)(n_x / 4));
    transpose_f16_kernel<<<(int)((n_wq + 255) / 256), 256, 0, stream>>>(Wqkv, wqT,
                                                                        D_MODEL, QKV_COLS);
    transpose_f16_kernel<<<(int)((n_wo + 255) / 256), 256, 0, stream>>>(Wo, woT,
                                                                        D_MODEL, D_MODEL);

    qkv_rope_kernel<<<dim3(QKV_COLS / 64, (B_SZ * T_SEQ) / 64), dim3(128), 0, stream>>>(
        xh, wqT, bqkv, qh, kh, vT);

    attention_kernel<<<dim3(T_SEQ / 16, B_SZ * N_HEADS), dim3(32), 0, stream>>>(
        qh, kh, vT, yh);

    outproj_kernel<<<dim3(D_MODEL / 64, (B_SZ * T_SEQ) / 64), dim3(128), 0, stream>>>(
        yh, woT, bo, out);
}
